// NestedResidualGIN_71373766525520
// MI455X (gfx1250) — matrix operand order repaired and checked by reading, not compiled, not run on hardware
//
#include <hip/hip_runtime.h>
#include <hip/hip_bf16.h>

// CDNA5 wave32 WMMA types
typedef __attribute__((ext_vector_type(16))) __bf16 v16bf;
typedef __attribute__((ext_vector_type(8)))  float  v8f;

#define NNODES   50000
#define NROWS    (NNODES * 4)      // 200000 rows of dim 64
#define NEDGES   800000
#define NGRAPHS  128
#define LN_EPS   1e-5f

// ---------------------------------------------------------------------------
// LDS fragment helpers (force ds_load_b128 via uint4 punning; all offsets are
// 16B aligned: row strides 72/200 bf16 = 144/400 B, column offsets mult of 8)
// ---------------------------------------------------------------------------
__device__ __forceinline__ v16bf ld_frag(const __bf16* p0, const __bf16* p1) {
    v16bf t;
    ((uint4*)&t)[0] = *(const uint4*)p0;
    ((uint4*)&t)[1] = *(const uint4*)p1;
    return t;
}
__device__ __forceinline__ void ld_row16_f32(const __bf16* p, float* dst) {
    v16bf t;
    ((uint4*)&t)[0] = ((const uint4*)p)[0];
    ((uint4*)&t)[1] = ((const uint4*)p)[1];
    #pragma unroll
    for (int j = 0; j < 16; ++j) dst[j] = (float)t[j];
}
__device__ __forceinline__ void st_row16_bf16(__bf16* p, const float* src) {
    v16bf t;
    #pragma unroll
    for (int j = 0; j < 16; ++j) t[j] = (__bf16)src[j];
    ((uint4*)p)[0] = ((uint4*)&t)[0];
    ((uint4*)p)[1] = ((uint4*)&t)[1];
}

// ---------------------------------------------------------------------------
// small utility kernels
// ---------------------------------------------------------------------------
__global__ void k_zero(float* __restrict__ p, int n) {
    int i = blockIdx.x * 256 + threadIdx.x;
    if (i < n) p[i] = 0.0f;
}

// pack Wq|Wk|Wv ([L][64][4][16] each) TRANSPOSED into bf16 [L][192][64] (n,k)
__global__ void k_pack_qkv(const float* __restrict__ Wq, const float* __restrict__ Wk,
                           const float* __restrict__ Wv, __bf16* __restrict__ out) {
    int i = blockIdx.x * 256 + threadIdx.x;
    if (i >= 3 * 192 * 64) return;
    int l = i / 12288, rem = i % 12288;
    int n = rem / 64,   d  = rem % 64;
    float v;
    if (n < 64)       v = Wq[l * 4096 + d * 64 + n];
    else if (n < 128) v = Wk[l * 4096 + d * 64 + (n - 64)];
    else              v = Wv[l * 4096 + d * 64 + (n - 128)];
    out[i] = (__bf16)v;
}

// transpose+convert: src [L][R][C] f32 -> dst [L][C][R] bf16
__global__ void k_cvt_t(const float* __restrict__ s, __bf16* __restrict__ d,
                        int L, int R, int C) {
    int i = blockIdx.x * 256 + threadIdx.x;
    if (i >= L * R * C) return;
    int l = i / (R * C), rem = i % (R * C);
    int c = rem / R,     r   = rem % R;
    d[i] = (__bf16)s[l * R * C + r * C + c];
}

__global__ void k_copy4(const float4* __restrict__ s, float4* __restrict__ d, int n4) {
    int i = blockIdx.x * 256 + threadIdx.x;
    if (i < n4) d[i] = s[i];
}

// ---------------------------------------------------------------------------
// edge scatter-add: agg[dst] += src_feat[src]   (256 floats per node)
// ---------------------------------------------------------------------------
__global__ __launch_bounds__(256) void k_scatter(const float* __restrict__ src_feat,
                                                 const int* __restrict__ src,
                                                 const int* __restrict__ dst,
                                                 float* __restrict__ agg) {
    long long gid = (long long)blockIdx.x * 256 + threadIdx.x;
    int e = (int)(gid >> 6);
    if (e >= NEDGES) return;
    int c = ((int)gid & 63) * 4;
    const float4 v = *(const float4*)(src_feat + (size_t)src[e] * 256 + c);
    float* p = agg + (size_t)dst[e] * 256 + c;
    atomicAdd(p + 0, v.x); atomicAdd(p + 1, v.y);
    atomicAdd(p + 2, v.z); atomicAdd(p + 3, v.w);
}

// ---------------------------------------------------------------------------
// wave-per-row LN / ReLU / pooling kernel (rows of 64 f32)
// ---------------------------------------------------------------------------
__global__ __launch_bounds__(256) void k_row64(const float* __restrict__ x,
                                               const float* __restrict__ g,
                                               const float* __restrict__ b,
                                               float* __restrict__ out,
                                               float* __restrict__ pool,
                                               const int* __restrict__ gids,
                                               int rows, int do_relu, int do_ln) {
    int row = blockIdx.x * 8 + (threadIdx.x >> 5);
    if (row >= rows) return;
    int lane = threadIdx.x & 31;
    size_t base = (size_t)row * 64;
    float x0 = x[base + lane], x1 = x[base + lane + 32];
    if (do_relu) { x0 = fmaxf(x0, 0.f); x1 = fmaxf(x1, 0.f); }
    float y0 = x0, y1 = x1;
    if (do_ln) {
        float s = x0 + x1, q = x0 * x0 + x1 * x1;
        #pragma unroll
        for (int off = 16; off > 0; off >>= 1) {
            s += __shfl_xor(s, off, 32);
            q += __shfl_xor(q, off, 32);
        }
        float mu = s * (1.f / 64.f);
        float rs = rsqrtf(q * (1.f / 64.f) - mu * mu + LN_EPS);
        y0 = (x0 - mu) * rs * g[lane]      + b[lane];
        y1 = (x1 - mu) * rs * g[lane + 32] + b[lane + 32];
    }
    if (out) { out[base + lane] = y0; out[base + lane + 32] = y1; }
    if (pool) {
        int node = row >> 2, sidx = row & 3;
        size_t pb = ((size_t)gids[node] * 4 + sidx) * 64;
        atomicAdd(pool + pb + lane,      y0);
        atomicAdd(pool + pb + lane + 32, y1);
    }
}

// ---------------------------------------------------------------------------
// Fused SAB block: LN -> QKV GEMM (WMMA bf16) -> softmax attention ->
//                  output GEMM (WMMA bf16) -> residual.
// Block = 128 threads (4 waves), 16 nodes (64 rows of dim 64).
// Weights arrive pre-transposed: Wqkv_T [192][64] (n,k), Wo_T [64][64] (n,k).
// ---------------------------------------------------------------------------
__global__ __launch_bounds__(128) void k_sab(const float* __restrict__ agg,
                                             const __bf16* __restrict__ WqkvT, // [192][64]
                                             const __bf16* __restrict__ WoT,   // [64][64]
                                             const float* __restrict__ lng,
                                             const float* __restrict__ lnb,
                                             const float* __restrict__ hres,   // null layer 0
                                             float* __restrict__ hout) {
    __shared__ __bf16 sX[64][72];      // LN'd input; later attention output
    __shared__ __bf16 sWt[192][72];    // W^T (n,k); later Wo^T in rows 0..63
    __shared__ __bf16 sQKV[64][200];   // projected q|k|v

    const int t    = threadIdx.x;
    const int wave = t >> 5;
    const int lane = t & 31;
    const int l15  = lane & 15;
    const int half = lane >> 4;        // 0: lanes 0-15, 1: lanes 16-31
    const int row0 = blockIdx.x * 64;

    // residual rows are first used at the very end -> prefetch now
    if (hres && t < 64)
        __builtin_prefetch(hres + (size_t)(row0 + t) * 64, 0, 3);

    // ---- phase 1: LN into bf16 (threads 0-63) + load W^T (threads 64-127)
    if (t < 64) {
        const float4* xr4 = (const float4*)(agg + (size_t)(row0 + t) * 64);
        float xv[64];
        float s = 0.f, q = 0.f;
        #pragma unroll
        for (int c4 = 0; c4 < 16; ++c4) {
            float4 v = xr4[c4];
            xv[c4 * 4 + 0] = v.x; xv[c4 * 4 + 1] = v.y;
            xv[c4 * 4 + 2] = v.z; xv[c4 * 4 + 3] = v.w;
            s += v.x + v.y + v.z + v.w;
            q += v.x * v.x + v.y * v.y + v.z * v.z + v.w * v.w;
        }
        float mu = s * (1.f / 64.f);
        float rs = rsqrtf(q * (1.f / 64.f) - mu * mu + LN_EPS);
        #pragma unroll
        for (int c = 0; c < 64; ++c)
            sX[t][c] = (__bf16)((xv[c] - mu) * rs * lng[c] + lnb[c]);
    } else {
        // 192 rows x 8 uint4 per row = 1536 vector copies over 64 threads
        for (int i = t - 64; i < 192 * 8; i += 64) {
            int r = i >> 3, qq = i & 7;
            ((uint4*)&sWt[r][0])[qq] = ((const uint4*)(WqkvT + r * 64))[qq];
        }
    }
    __syncthreads();

    // ---- phase 2: qkv = sX[64x64] @ W[64x192]  (per wave: one 16-row M tile)
    {
        const int m = wave * 16 + l15;
        const int kb = half * 8;
        // A fragments are invariant across the 12 N tiles: load once
        const v16bf a0 = ld_frag(&sX[m][kb],      &sX[m][kb + 16]);
        const v16bf a1 = ld_frag(&sX[m][32 + kb], &sX[m][48 + kb]);
        for (int nt = 0; nt < 12; ++nt) {
            const int n = nt * 16 + l15;
            const int ks = half * 8;
            v16bf b0 = ld_frag(&sWt[n][ks],      &sWt[n][ks + 16]);
            v16bf b1 = ld_frag(&sWt[n][32 + ks], &sWt[n][48 + ks]);
            v8f acc = {};
            acc = __builtin_amdgcn_wmma_f32_16x16x32_bf16(false, a0, false, b0,
                                                          (short)0, acc, false, false);
            acc = __builtin_amdgcn_wmma_f32_16x16x32_bf16(false, a1, false, b1,
                                                          (short)0, acc, false, false);
            #pragma unroll
            for (int r = 0; r < 8; ++r)
                sQKV[wave * 16 + half * 8 + r][nt * 16 + l15] = (__bf16)acc[r];
        }
    }
    __syncthreads();

    // ---- phase 3: per-(node,head) attention (threads 0-63) + load Wo^T (64-127)
    if (t < 64) {
        const int nl = t >> 2, hd = t & 3;
        const int r0 = nl * 4, qc = hd * 16, kc = 64 + hd * 16, vc = 128 + hd * 16;
        float kv[4][16];                      // K rows, later reused for V rows
        #pragma unroll
        for (int sk = 0; sk < 4; ++sk) ld_row16_f32(&sQKV[r0 + sk][kc], kv[sk]);
        float sc[4][4];
        #pragma unroll
        for (int sq = 0; sq < 4; ++sq) {
            float qv[16];
            ld_row16_f32(&sQKV[r0 + sq][qc], qv);
            #pragma unroll
            for (int sk = 0; sk < 4; ++sk) {
                float s = 0.f;
                #pragma unroll
                for (int k = 0; k < 16; ++k) s += qv[k] * kv[sk][k];
                sc[sq][sk] = s * 0.25f;       // 1/sqrt(16)
            }
        }
        #pragma unroll
        for (int sq = 0; sq < 4; ++sq) {
            float mx = fmaxf(fmaxf(sc[sq][0], sc[sq][1]), fmaxf(sc[sq][2], sc[sq][3]));
            float den = 0.f;
            #pragma unroll
            for (int sk = 0; sk < 4; ++sk) { sc[sq][sk] = __expf(sc[sq][sk] - mx); den += sc[sq][sk]; }
            float inv = 1.f / den;
            #pragma unroll
            for (int sk = 0; sk < 4; ++sk) sc[sq][sk] *= inv;
        }
        #pragma unroll
        for (int sk = 0; sk < 4; ++sk) ld_row16_f32(&sQKV[r0 + sk][vc], kv[sk]);
        #pragma unroll
        for (int sq = 0; sq < 4; ++sq) {
            float o[16];
            #pragma unroll
            for (int k = 0; k < 16; ++k) o[k] = 0.f;
            #pragma unroll
            for (int sk = 0; sk < 4; ++sk) {
                float w = sc[sq][sk];
                #pragma unroll
                for (int k = 0; k < 16; ++k) o[k] += w * kv[sk][k];
            }
            st_row16_bf16(&sX[r0 + sq][qc], o);   // reuse sX for attention output
        }
    } else {
        for (int i = t - 64; i < 64 * 8; i += 64) {
            int r = i >> 3, qq = i & 7;
            ((uint4*)&sWt[r][0])[qq] = ((const uint4*)(WoT + r * 64))[qq];
        }
    }
    __syncthreads();

    // ---- phase 4: proj = attn_out[64x64] @ Wo[64x64], epilogue residual
    {
        const int m = wave * 16 + l15;
        const int kb = half * 8;
        const v16bf a0 = ld_frag(&sX[m][kb],      &sX[m][kb + 16]);
        const v16bf a1 = ld_frag(&sX[m][32 + kb], &sX[m][48 + kb]);
        for (int nt = 0; nt < 4; ++nt) {
            const int n = nt * 16 + l15;
            const int ks = half * 8;
            v16bf b0 = ld_frag(&sWt[n][ks],      &sWt[n][ks + 16]);
            v16bf b1 = ld_frag(&sWt[n][32 + ks], &sWt[n][48 + ks]);
            v8f acc = {};
            acc = __builtin_amdgcn_wmma_f32_16x16x32_bf16(false, a0, false, b0,
                                                          (short)0, acc, false, false);
            acc = __builtin_amdgcn_wmma_f32_16x16x32_bf16(false, a1, false, b1,
                                                          (short)0, acc, false, false);
            #pragma unroll
            for (int r = 0; r < 8; ++r) {
                const int lr  = wave * 16 + half * 8 + r;
                const int col = nt * 16 + l15;
                size_t idx = (size_t)(row0 + lr) * 64 + col;
                float v = acc[r];
                if (hres) v += hres[idx] + agg[idx];   // h + agg + proj (layers 1,2)
                hout[idx] = v;
            }
        }
    }
}

// ---------------------------------------------------------------------------
// score += pooled[512,64] @ W[64,16] + bias[16]   (WMMA, 8 blocks of 64 rows)
// W arrives transposed: [16][64] (n,k)
// ---------------------------------------------------------------------------
__global__ __launch_bounds__(128) void k_pred(const float* __restrict__ pooled,
                                              const __bf16* __restrict__ WT,  // [16][64]
                                              const float* __restrict__ bias,
                                              float* __restrict__ out) {
    __shared__ __bf16 sA[64][72];
    __shared__ __bf16 sBt[16][72];
    const int t = threadIdx.x;
    const int wave = t >> 5, lane = t & 31, l15 = lane & 15, half = lane >> 4;
    const int row0 = blockIdx.x * 64;

    for (int i = t; i < 64 * 64; i += 128)
        sA[i >> 6][i & 63] = (__bf16)pooled[(size_t)row0 * 64 + i];
    if (t < 16 * 8) {                        // 16 rows x 8 uint4
        int r = t >> 3, qq = t & 7;
        ((uint4*)&sBt[r][0])[qq] = ((const uint4*)(WT + r * 64))[qq];
    }
    __syncthreads();

    const int m = wave * 16 + l15;
    const int kb = half * 8, ks = half * 16;
    const v16bf a0 = ld_frag(&sA[m][kb],      &sA[m][kb + 16]);
    const v16bf a1 = ld_frag(&sA[m][32 + kb], &sA[m][48 + kb]);
    const v16bf b0 = ld_frag(&sBt[l15][kb],      &sBt[l15][kb + 16]);
    const v16bf b1 = ld_frag(&sBt[l15][32 + kb], &sBt[l15][48 + kb]);
    v8f acc = {};
    acc = __builtin_amdgcn_wmma_f32_16x16x32_bf16(false, a0, false, b0,
                                                  (short)0, acc, false, false);
    acc = __builtin_amdgcn_wmma_f32_16x16x32_bf16(false, a1, false, b1,
                                                  (short)0, acc, false, false);
    #pragma unroll
    for (int r = 0; r < 8; ++r) {
        const int row = row0 + wave * 16 + half * 8 + r;
        out[row * 16 + l15] += acc[r] + bias[l15];
    }
}

// ---------------------------------------------------------------------------
extern "C" void kernel_launch(void* const* d_in, const int* in_sizes, int n_in,
                              void* d_out, int out_size, void* d_ws, size_t ws_size,
                              hipStream_t stream) {
    const float* feat  = (const float*)d_in[0];
    const int*   src   = (const int*)d_in[1];
    const int*   dst   = (const int*)d_in[2];
    const int*   gids  = (const int*)d_in[3];
    const float* Wq    = (const float*)d_in[4];
    const float* Wk    = (const float*)d_in[5];
    const float* Wv    = (const float*)d_in[6];
    const float* Wo    = (const float*)d_in[7];
    const float* sabg  = (const float*)d_in[8];
    const float* sabb  = (const float*)d_in[9];
    const float* ln1g  = (const float*)d_in[10];
    const float* ln1b  = (const float*)d_in[11];
    const float* ln2g  = (const float*)d_in[12];
    const float* ln2b  = (const float*)d_in[13];
    const float* Wpred = (const float*)d_in[14];
    const float* bpred = (const float*)d_in[15];
    float* out = (float*)d_out;

    // workspace layout (all L2-resident: ~154 MB)
    float* h      = (float*)d_ws;
    float* hpre   = h    + (size_t)NROWS * 64;
    float* agg    = hpre + (size_t)NROWS * 64;
    float* pooled = agg  + (size_t)NROWS * 64;               // 512*64 f32
    __bf16* Wqkv_bf  = (__bf16*)(pooled + 512 * 64);         // 3*192*64 (transposed)
    __bf16* Wo_bf    = Wqkv_bf + 3 * 192 * 64;               // 3*64*64  (transposed)
    __bf16* Wpred_bf = Wo_bf   + 3 * 4096;                   // 4*16*64  (transposed)

    // weight conversion (transposed to [n][k] for vectorized B-fragment loads)
    k_pack_qkv<<<(3 * 192 * 64 + 255) / 256, 256, 0, stream>>>(Wq, Wk, Wv, Wqkv_bf);
    k_cvt_t<<<(3 * 64 * 64 + 255) / 256, 256, 0, stream>>>(Wo, Wo_bf, 3, 64, 64);
    k_cvt_t<<<(4 * 64 * 16 + 255) / 256, 256, 0, stream>>>(Wpred, Wpred_bf, 4, 64, 16);
    k_zero<<<(8192 + 255) / 256, 256, 0, stream>>>(out, 8192);

    const int lnBlocks = (NROWS + 7) / 8;   // 25000

    // hidden_rep[0] = feat: pool + prediction term 0
    k_zero<<<(512 * 64 + 255) / 256, 256, 0, stream>>>(pooled, 512 * 64);
    k_row64<<<lnBlocks, 256, 0, stream>>>(feat, nullptr, nullptr, nullptr, pooled,
                                          gids, NROWS, 0, 0);
    k_pred<<<8, 128, 0, stream>>>(pooled, Wpred_bf, bpred, out);

    const float* gatherSrc = feat;
    for (int l = 0; l < 3; ++l) {
        if (l > 0) {
            // h_pre = LN1(h)
            k_row64<<<lnBlocks, 256, 0, stream>>>(h, ln1g + l * 64, ln1b + l * 64,
                                                  hpre, nullptr, nullptr, NROWS, 0, 1);
            gatherSrc = hpre;
        }
        // agg = gatherSrc + segment_sum(gatherSrc[src], dst)
        k_copy4<<<(NROWS * 16 + 255) / 256, 256, 0, stream>>>((const float4*)gatherSrc,
                                                              (float4*)agg, NROWS * 16);
        k_scatter<<<(NEDGES * 64) / 256, 256, 0, stream>>>(gatherSrc, src, dst, agg);
        // fused SAB (layer 0: h = proj ; layers 1,2: h = h + agg + proj)
        k_sab<<<NROWS / 64, 128, 0, stream>>>(agg, Wqkv_bf + l * 192 * 64,
                                              Wo_bf + l * 4096,
                                              sabg + l * 64, sabb + l * 64,
                                              (l == 0) ? nullptr : h, h);
        // hidden_rep[l+1] = LN2(relu(h)): pool + prediction term
        k_zero<<<(512 * 64 + 255) / 256, 256, 0, stream>>>(pooled, 512 * 64);
        k_row64<<<lnBlocks, 256, 0, stream>>>(h, ln2g + l * 64, ln2b + l * 64,
                                              nullptr, pooled, gids, NROWS, 1, 1);
        k_pred<<<8, 128, 0, stream>>>(pooled, Wpred_bf + (l + 1) * 1024,
                                      bpred + (l + 1) * 16, out);
    }
}
